// LightSelfAttention_75265006895266
// MI455X (gfx1250) — matrix operand order repaired and checked by reading, not compiled, not run on hardware
//
#include <hip/hip_runtime.h>

// Problem dims (fixed by the reference)
#define Bb 8
#define Cc 128
#define NN 4096           // H*W = 64*64
// 1/sqrt(128) * log2(e): Q is pre-scaled so softmax can use native exp2.
#define QSCALE 0.12751591807079106f

typedef __attribute__((ext_vector_type(16))) __bf16 v16bf;
typedef __attribute__((ext_vector_type(8)))  float  v8f;

union BF16Frag {
    v16bf v;
    uint4 q[2];
    unsigned short h[16];
};

// Native fp32->bf16 conversions (backend picks v_cvt[_pk]_bf16_f32 if present)
__device__ inline unsigned short f2bf(float f) {
    union { __bf16 b; unsigned short u; } c;
    c.b = (__bf16)f;
    return c.u;
}
__device__ inline unsigned int pack2bf(float lo, float hi) {
    union { __bf16 b[2]; unsigned int u; } c;
    c.b[0] = (__bf16)lo;
    c.b[1] = (__bf16)hi;
    return c.u;
}

__device__ inline v8f vzero8() {
    v8f z;
#pragma unroll
    for (int i = 0; i < 8; ++i) z[i] = 0.f;
    return z;
}

// -----------------------------------------------------------------------------
// Kernel 0: one-shot conversion of Wq|Wk|Wv (fp32) -> bf16 in workspace.
// -----------------------------------------------------------------------------
__global__ __launch_bounds__(256) void wconv_kernel(
    const float* __restrict__ Wq, const float* __restrict__ Wk,
    const float* __restrict__ Wv, unsigned short* __restrict__ Wb)
{
    int idx = blockIdx.x * 256 + threadIdx.x;        // 0 .. 3*128*128-1
    if (idx >= 3 * Cc * Cc) return;
    int p = idx / (Cc * Cc), r = idx % (Cc * Cc);
    const float* W = (p == 0) ? Wq : (p == 1) ? Wk : Wv;
    Wb[idx] = f2bf(W[r]);
}

// -----------------------------------------------------------------------------
// One projection with compile-time output layout.
//   P=0: Qt[n][c] pre-scaled by QSCALE   (packed b128 stores)
//   P=1: Kt[m][c]                        (packed b128 stores)
//   P=2: Vc[c][m]                        (immediate-offset b16 stores)
// -----------------------------------------------------------------------------
template<int P>
__device__ __forceinline__ void do_proj(
    const unsigned short* __restrict__ Wb,
    const float* __restrict__ bias,
    unsigned short* __restrict__ dst,            // batch-local base
    const unsigned short (*xT)[Cc + 8],
    int n0, int o0, int lrow, int half)
{
    v8f acc[8];
#pragma unroll
    for (int t = 0; t < 8; ++t) acc[t] = vzero8();

#pragma unroll
    for (int kk = 0; kk < 4; ++kk) {             // K-steps over C (32 each)
        BF16Frag a;
        const unsigned short* wrow =
            Wb + (size_t)P * Cc * Cc + (size_t)(o0 + lrow) * Cc
               + 32 * kk + 16 * half;
        a.q[0] = *(const uint4*)(wrow);
        a.q[1] = *(const uint4*)(wrow + 8);

        BF16Frag bf[8];
#pragma unroll
        for (int t = 0; t < 8; ++t) {
            const unsigned short* src = &xT[t * 16 + lrow][32 * kk + 16 * half];
            bf[t].q[0] = *(const uint4*)(src);
            bf[t].q[1] = *(const uint4*)(src + 8);
        }
#pragma unroll
        for (int t = 0; t < 8; ++t)
            acc[t] = __builtin_amdgcn_wmma_f32_16x16x32_bf16(
                false, a.v, false, bf[t].v, (short)0, acc[t], false, false);
    }

    float bvals[8];
#pragma unroll
    for (int r = 0; r < 8; ++r) bvals[r] = bias[o0 + 8 * half + r];

    if (P <= 1) {
        // Lane's rows r=0..7 are 8 consecutive bf16 at [n][o0+8*half .. +8].
        const float sc = (P == 0) ? QSCALE : 1.0f;
#pragma unroll
        for (int t = 0; t < 8; ++t) {
            int n = n0 + t * 16 + lrow;
            uint4 pk;
            pk.x = pack2bf((acc[t][0] + bvals[0]) * sc, (acc[t][1] + bvals[1]) * sc);
            pk.y = pack2bf((acc[t][2] + bvals[2]) * sc, (acc[t][3] + bvals[3]) * sc);
            pk.z = pack2bf((acc[t][4] + bvals[4]) * sc, (acc[t][5] + bvals[5]) * sc);
            pk.w = pack2bf((acc[t][6] + bvals[6]) * sc, (acc[t][7] + bvals[7]) * sc);
            *(uint4*)(dst + (size_t)n * Cc + o0 + 8 * half) = pk;
        }
    } else {
        // V: channel-major; (r,t) offsets are compile-time immediates.
        unsigned short* vdst = dst + (size_t)(o0 + 8 * half) * NN + n0 + lrow;
#pragma unroll
        for (int t = 0; t < 8; ++t) {
#pragma unroll
            for (int r = 0; r < 8; ++r)
                vdst[(size_t)r * NN + t * 16] = f2bf(acc[t][r] + bvals[r]);
        }
    }
}

// -----------------------------------------------------------------------------
// Kernel 1: QKV projection. Grid: Bb*(NN/128) blocks, 256 thr (8 waves).
// -----------------------------------------------------------------------------
__global__ __launch_bounds__(256) void qkv_kernel(
    const float* __restrict__ x,
    const unsigned short* __restrict__ Wb,
    const float* __restrict__ bq, const float* __restrict__ bk,
    const float* __restrict__ bv,
    unsigned short* __restrict__ Qt,
    unsigned short* __restrict__ Kt,
    unsigned short* __restrict__ Vc)
{
    __shared__ unsigned short xT[128][Cc + 8];   // [n_local][c], bf16, padded

    const int b   = blockIdx.x >> 5;
    const int n0  = (blockIdx.x & 31) * 128;
    const int tid = threadIdx.x;

    for (int idx = tid; idx < 128 * Cc; idx += 256) {
        int c = idx >> 7;
        int n = idx & 127;
        xT[n][c] = f2bf(x[((size_t)b * Cc + c) * NN + (n0 + n)]);
    }
    __syncthreads();

    const int wave = tid >> 5, lane = tid & 31;
    const int lrow = lane & 15, half = lane >> 4;
    const int o0   = wave * 16;

    do_proj<0>(Wb, bq, Qt + (size_t)b * NN * Cc, xT, n0, o0, lrow, half);
    __syncthreads();   // fence: keeps per-projection LDS reads from merging
    do_proj<1>(Wb, bk, Kt + (size_t)b * NN * Cc, xT, n0, o0, lrow, half);
    __syncthreads();
    do_proj<2>(Wb, bv, Vc + (size_t)b * Cc * NN, xT, n0, o0, lrow, half);
}

// -----------------------------------------------------------------------------
// Fragment group loaders (8 tiles of one 32-wide k-step)
// -----------------------------------------------------------------------------
__device__ __forceinline__ void load_kgrp(
    BF16Frag (&kf)[8], const unsigned short* kbase, int m0, int kk)
{
#pragma unroll
    for (int t = 0; t < 8; ++t) {
        const unsigned short* krow = kbase + (size_t)(m0 + t * 16) * Cc + 32 * kk;
        kf[t].q[0] = *(const uint4*)(krow);
        kf[t].q[1] = *(const uint4*)(krow + 8);
    }
}

__device__ __forceinline__ void load_vgrp(
    BF16Frag (&vf)[8], const unsigned short* vbase, int m0, int kk)
{
#pragma unroll
    for (int ct = 0; ct < 8; ++ct) {
        const unsigned short* vrow = vbase + (size_t)(ct * 16) * NN + m0 + kk * 32;
        vf[ct].q[0] = *(const uint4*)(vrow);
        vf[ct].q[1] = *(const uint4*)(vrow + 8);
    }
}

// -----------------------------------------------------------------------------
// Kernel 2: fused flash attention + residual.
//   S = Q^T K (Q pre-scaled by 1/sqrt(C)*log2e) ; P = softmax (exp2 domain) ;
//   O^T = P V^T ; out = O + x
// Double-buffered K/V fragment groups: loads for k-step kk+1 are issued while
// the 8 WMMAs of k-step kk execute; first V group issues before the softmax so
// its global latency hides behind the exp2/shuffle work.
// -----------------------------------------------------------------------------
__global__ __launch_bounds__(256) void attn_kernel(
    const unsigned short* __restrict__ Qt,
    const unsigned short* __restrict__ Kt,
    const unsigned short* __restrict__ Vc,
    const float* __restrict__ x,
    float* __restrict__ out)
{
    const int b    = blockIdx.x >> 5;
    const int n0   = (blockIdx.x & 31) * 128;
    const int wave = threadIdx.x >> 5, lane = threadIdx.x & 31;
    const int lrow = lane & 15, half = lane >> 4;
    const int nrow0 = n0 + wave * 16;

    // Per-wave P staging: 16 rows x 128 cols bf16 (stride 136 halfs = 272B,
    // multiple of 16B -> aligned b128 LDS reads).
    __shared__ unsigned short Plds[8][16][136];

    // Resident Q fragments (4 K-steps over C=128).
    BF16Frag qf[4];
    {
        const unsigned short* qrow =
            Qt + ((size_t)b * NN + (nrow0 + lrow)) * Cc;
#pragma unroll
        for (int kk = 0; kk < 4; ++kk) {
            qf[kk].q[0] = *(const uint4*)(qrow + 32 * kk + 16 * half);
            qf[kk].q[1] = *(const uint4*)(qrow + 32 * kk + 8 + 16 * half);
        }
    }

    v8f Oacc[8];
#pragma unroll
    for (int t = 0; t < 8; ++t) Oacc[t] = vzero8();
    float m_row[8], l_row[8];
#pragma unroll
    for (int r = 0; r < 8; ++r) { m_row[r] = -1e30f; l_row[r] = 0.f; }

    const unsigned short* kbase =
        Kt + ((size_t)b * NN + lrow) * Cc + 16 * half;
    const unsigned short* vbase =
        Vc + ((size_t)b * Cc + lrow) * NN + 16 * half;

#pragma unroll 1
    for (int m0 = 0; m0 < NN; m0 += 128) {
        // Prefetch next key block's K and V lines (global_prefetch_b8).
        if (m0 + 128 < NN) {
            const char* kpre = (const char*)(Kt + ((size_t)b * NN + (m0 + 128)) * Cc);
#pragma unroll
            for (int i = 0; i < 4; ++i)
                __builtin_prefetch(kpre + ((size_t)(i * 32 + lane)) * 128, 0, 3);
#pragma unroll
            for (int i = 0; i < 4; ++i)
                __builtin_prefetch(
                    (const char*)(Vc + ((size_t)b * Cc + (i * 32 + lane)) * NN
                                     + (m0 + 128)), 0, 3);
        }

        // ---- S tiles: 8 tiles of 16x16; double-buffered K fragments ----
        v8f S[8];
#pragma unroll
        for (int t = 0; t < 8; ++t) S[t] = vzero8();

        BF16Frag kf[2][8];
        load_kgrp(kf[0], kbase, m0, 0);
#pragma unroll
        for (int kk = 0; kk < 4; ++kk) {
            if (kk < 3) load_kgrp(kf[(kk + 1) & 1], kbase, m0, kk + 1);
#pragma unroll
            for (int t = 0; t < 8; ++t)
                S[t] = __builtin_amdgcn_wmma_f32_16x16x32_bf16(
                    false, qf[kk].v, false, kf[kk & 1][t].v, (short)0,
                    S[t], false, false);
        }

        // Issue first V fragment group now: latency hides behind softmax.
        BF16Frag vf[2][8];
        load_vgrp(vf[0], vbase, m0, 0);

        // ---- online softmax (row r lives in 16-lane halves of VGPR r) ----
        float mnew[8];
#pragma unroll
        for (int r = 0; r < 8; ++r) {
            float mx = m_row[r];
#pragma unroll
            for (int t = 0; t < 8; ++t) mx = fmaxf(mx, S[t][r]);
            mx = fmaxf(mx, __shfl_xor(mx, 1, 32));
            mx = fmaxf(mx, __shfl_xor(mx, 2, 32));
            mx = fmaxf(mx, __shfl_xor(mx, 4, 32));
            mx = fmaxf(mx, __shfl_xor(mx, 8, 32));
            mnew[r] = mx;
        }

#pragma unroll
        for (int r = 0; r < 8; ++r) {
            float alpha = exp2f(m_row[r] - mnew[r]);   // native v_exp_f32
            float rs = 0.f;
#pragma unroll
            for (int t = 0; t < 8; ++t) {
                float pv = exp2f(S[t][r] - mnew[r]);
                rs += pv;
                *(__bf16*)&Plds[wave][r + 8 * half][t * 16 + lrow] = (__bf16)pv;
            }
            rs += __shfl_xor(rs, 1, 32);
            rs += __shfl_xor(rs, 2, 32);
            rs += __shfl_xor(rs, 4, 32);
            rs += __shfl_xor(rs, 8, 32);
            l_row[r] = l_row[r] * alpha + rs;
            m_row[r] = mnew[r];
#pragma unroll
            for (int ct = 0; ct < 8; ++ct) Oacc[ct][r] *= alpha;
        }

        // ---- O += P * V^T : double-buffered V fragments ----
#pragma unroll
        for (int kk = 0; kk < 4; ++kk) {         // K-steps over 128 keys
            if (kk < 3) load_vgrp(vf[(kk + 1) & 1], vbase, m0, kk + 1);

            BF16Frag pf;
            const unsigned short* prow = &Plds[wave][lrow][kk * 32 + 16 * half];
            pf.q[0] = *(const uint4*)(prow);
            pf.q[1] = *(const uint4*)(prow + 8);

#pragma unroll
            for (int ct = 0; ct < 8; ++ct)
                Oacc[ct] = __builtin_amdgcn_wmma_f32_16x16x32_bf16(
                    false, pf.v, false, vf[kk & 1][ct].v, (short)0,
                    Oacc[ct], false, false);
        }
    }

    // ---- epilogue: rows r=0..7 are 8 consecutive n -> dense float4 I/O ----
    float inv[8];
#pragma unroll
    for (int r = 0; r < 8; ++r) inv[r] = 1.0f / l_row[r];

#pragma unroll
    for (int ct = 0; ct < 8; ++ct) {
        int c = ct * 16 + lrow;
        size_t base = ((size_t)b * Cc + c) * NN + (size_t)nrow0 + 8 * half;
        float4 x0 = *(const float4*)(x + base);
        float4 x1 = *(const float4*)(x + base + 4);
        float4 o0, o1;
        o0.x = Oacc[ct][0] * inv[0] + x0.x;
        o0.y = Oacc[ct][1] * inv[1] + x0.y;
        o0.z = Oacc[ct][2] * inv[2] + x0.z;
        o0.w = Oacc[ct][3] * inv[3] + x0.w;
        o1.x = Oacc[ct][4] * inv[4] + x1.x;
        o1.y = Oacc[ct][5] * inv[5] + x1.y;
        o1.z = Oacc[ct][6] * inv[6] + x1.z;
        o1.w = Oacc[ct][7] * inv[7] + x1.w;
        *(float4*)(out + base)     = o0;
        *(float4*)(out + base + 4) = o1;
    }
}

// -----------------------------------------------------------------------------
extern "C" void kernel_launch(void* const* d_in, const int* in_sizes, int n_in,
                              void* d_out, int out_size, void* d_ws, size_t ws_size,
                              hipStream_t stream)
{
    const float* x  = (const float*)d_in[0];
    const float* Wq = (const float*)d_in[1];
    const float* bq = (const float*)d_in[2];
    const float* Wk = (const float*)d_in[3];
    const float* bk = (const float*)d_in[4];
    const float* Wv = (const float*)d_in[5];
    const float* bv = (const float*)d_in[6];

    // Workspace: Qt | Kt | Vc (bf16, 8 MB each) | Wb (bf16, 96 KB)
    unsigned short* Qt = (unsigned short*)d_ws;
    unsigned short* Kt = Qt + (size_t)Bb * NN * Cc;
    unsigned short* Vc = Kt + (size_t)Bb * NN * Cc;
    unsigned short* Wb = Vc + (size_t)Bb * NN * Cc;

    wconv_kernel<<<(3 * Cc * Cc + 255) / 256, 256, 0, stream>>>(Wq, Wk, Wv, Wb);

    dim3 grid(Bb * (NN / 128));
    dim3 block(256);
    qkv_kernel<<<grid, block, 0, stream>>>(x, Wb, bq, bk, bv, Qt, Kt, Vc);
    attn_kernel<<<grid, block, 0, stream>>>(Qt, Kt, Vc, x, (float*)d_out);
}
